// Dynamics_40355512713600
// MI455X (gfx1250) — matrix operand order repaired and checked by reading, not compiled, not run on hardware
//
#include <hip/hip_runtime.h>
#include <hip/hip_bf16.h>
#include <math.h>

// Problem constants (from the reference)
#define DIM   8
#define HALFD 4
#define NS    1024
#define ND    6
#define BATCH 16384
#define TS    16          // samples per wave
#define NG    9           // groups: 0 = value (16 samples), 1..8 = jacobian (2 samples x 8 cols)
#define SCALE 0.044194173824159216f  // sqrt(2/1024)

typedef float    v2f  __attribute__((ext_vector_type(2)));
typedef float    v4f  __attribute__((ext_vector_type(4)));
typedef float    v8f  __attribute__((ext_vector_type(8)));
typedef _Float16 v2h  __attribute__((ext_vector_type(2)));
typedef _Float16 v16h __attribute__((ext_vector_type(16)));
typedef __fp16   v2fp16 __attribute__((ext_vector_type(2)));

union H16 { v16h v; v2h p[8]; uint4 q[2]; };

static __device__ inline v2h pk2(float a, float b) {
#if __has_builtin(__builtin_amdgcn_cvt_pkrtz)
    union { v2fp16 f; v2h h; } u;
    u.f = __builtin_amdgcn_cvt_pkrtz(a, b);   // v_cvt_pk_rtz_f16_f32
    return u.h;
#else
    v2h r; r.x = (_Float16)a; r.y = (_Float16)b; return r;
#endif
}

// One wave32 per 16 samples.
//  GEMM1 (W @ [x1|J1])  : fp32 WMMA 16x16x4 (K=4 == HALFD exactly), 100% N-util.
//  GEMM2 value column   : fp32 WMMA 16x16x4, K-chunks (w16+i, w16+i+1, w16+i+8, w16+i+9)
//                         so B-operand == pair of GEMM1 D registers (zero shuffles).
//  GEMM2 jacobian cols  : f16 WMMA 16x16x32 (8x FLOPs/instr, f32 accumulate); K order
//                         permuted so B-operand == cvt_pk of GEMM1 D register pairs
//                         and A-operand == two contiguous b128 LDS loads of f16 A.
__global__ __launch_bounds__(32)
void diffeo_dynamics_kernel(const float* __restrict__ x,
                            const float* __restrict__ Ws,
                            const float* __restrict__ bs,
                            const float* __restrict__ As,
                            float* __restrict__ out)
{
    __shared__ float Xs[TS][DIM];            // current x per sample
    __shared__ float Js[TS][DIM][DIM];       // current Jacobian per sample
    __shared__ float Sg[2][HALFD][TS][NG];   // [machine][row][sample][0=val,1..8=d/dx_c]
    __shared__ float snL[TS][16];            // -sin(u+b)*scale broadcast tile [sample][row]
    __shared__ _Float16 AfL[HALFD][NS];      // f16 copy of A for the jacobian GEMM2 (8KB)

    const int lane = threadIdx.x;            // 0..31 (wave32)
    const int half = lane >> 4;              // lane half <-> K rows {0,1} vs {2,3}
    const int l15  = lane & 15;
    const int base = blockIdx.x * TS;

    // ---- init state: Xs = x, Js = I ----
    #pragma unroll
    for (int t = 0; t < (TS * DIM) / 32; ++t) {
        int idx = t * 32 + lane;
        Xs[idx >> 3][idx & 7] = x[(size_t)(base + (idx >> 3)) * DIM + (idx & 7)];
    }
    #pragma unroll
    for (int t = 0; t < (TS * DIM * DIM) / 32; ++t) {
        int idx = t * 32 + lane;
        int s = idx >> 6, rc = idx & 63;
        Js[s][rc >> 3][rc & 7] = ((rc >> 3) == (rc & 7)) ? 1.0f : 0.0f;
    }
    __syncthreads();

    const v8f vzero = {0.f, 0.f, 0.f, 0.f, 0.f, 0.f, 0.f, 0.f};

    #pragma unroll 1
    for (int k = 0; k < ND; ++k) {
        const int h1 = (k & 1) ? HALFD : 0;  // conditioning half
        const int h2 = HALFD - h1;           // transformed half

        // ---- GEMM1 B-operands (4x16 Z tiles), shared by both machines ----
        v2f B1[NG];
        B1[0].x = Xs[l15][h1 + 2 * half + 0];          // value group: N = sample
        B1[0].y = Xs[l15][h1 + 2 * half + 1];
        #pragma unroll
        for (int g = 0; g < 8; ++g) {                  // jac groups: N = (sample pair, input col)
            int s = 2 * g + (l15 >> 3);
            int c = lane & 7;
            B1[g + 1].x = Js[s][h1 + 2 * half + 0][c];
            B1[g + 1].y = Js[s][h1 + 2 * half + 1][c];
        }

        #pragma unroll
        for (int m = 0; m < 2; ++m) {                  // m=0: s-net, m=1: t-net
            const float* W = Ws + (size_t)(k * 2 + m) * NS * HALFD;
            const float* b = bs + (size_t)(k * 2 + m) * NS;
            const float* A = As + (size_t)(k * 2 + m) * HALFD * NS;

            // ---- stage A as f16 in LDS (row-major [i][j]) ----
            __syncthreads();
            #pragma unroll 1
            for (int t = 0; t < (HALFD * NS) / 64; ++t) {     // 64 iters, 2 elems/lane
                int idx = (t * 32 + lane) * 2;
                v2f av = *(const v2f*)(A + idx);
                *(v2h*)&AfL[0][idx] = pk2(av.x, av.y);
            }
            __syncthreads();

            v8f acc[NG];
            #pragma unroll
            for (int g = 0; g < NG; ++g) acc[g] = vzero;

            #pragma unroll 1
            for (int w2 = 0; w2 < NS / 32; ++w2) {     // W-tile pairs (K=32 for f16 GEMM2)
                H16 Bh[8];
                #pragma unroll
                for (int p = 0; p < 2; ++p) {
                    const int w = w2 * 2 + p;

                    // GEMM1 A-operand: W rows w*16..w*16+15 (row-major 1024x4)
                    v2f a1 = *(const v2f*)(W + (size_t)(w * 16 + l15) * HALFD + 2 * half);

                    // value pass
                    v8f U0 = __builtin_amdgcn_wmma_f32_16x16x4_f32(
                        false, a1, false, B1[0], (short)0, vzero, false, false);

                    // biases for my rows {8*half .. 8*half+7}: broadcast b128 loads
                    v4f ba = *(const v4f*)(b + w * 16 + 8 * half);
                    v4f bb = *(const v4f*)(b + w * 16 + 8 * half + 4);
                    float sn[8];
                    #pragma unroll
                    for (int i = 0; i < 8; ++i) {
                        float bi = (i < 4) ? ba[i] : bb[i - 4];
                        float ub = U0[i] + bi;
                        sn[i] = __sinf(ub) * (-SCALE);
                        U0[i] = __cosf(ub) * SCALE;    // feat (value column stays fp32)
                    }
                    // publish -sin*scale per (sample, row) for the jac groups
                    v4f s0; s0[0]=sn[0]; s0[1]=sn[1]; s0[2]=sn[2]; s0[3]=sn[3];
                    v4f s1; s1[0]=sn[4]; s1[1]=sn[5]; s1[2]=sn[6]; s1[3]=sn[7];
                    *(v4f*)&snL[l15][8 * half]     = s0;
                    *(v4f*)&snL[l15][8 * half + 4] = s1;

                    // GEMM2 value column: fp32 K-chunks; B = pair of U0 registers
                    #pragma unroll
                    for (int q = 0; q < 4; ++q) {
                        int i = 2 * q;
                        v2f a2 = *(const v2f*)(A + (size_t)(lane & 3) * NS + w * 16 + i + 8 * half);
                        v2f bf; bf.x = U0[i]; bf.y = U0[i + 1];
                        acc[0] = __builtin_amdgcn_wmma_f32_16x16x4_f32(
                            false, a2, false, bf, (short)0, acc[0], false, false);
                    }

                    // jacobian groups: GEMM1 + pack dfeat into f16 B-operands
                    #pragma unroll
                    for (int g = 0; g < 8; ++g) {
                        v8f Ug = __builtin_amdgcn_wmma_f32_16x16x4_f32(
                            false, a1, false, B1[g + 1], (short)0, vzero, false, false);
                        int sIdx = 2 * g + (l15 >> 3);             // my sample for this group
                        v4f n0 = *(const v4f*)&snL[sIdx][8 * half];
                        v4f n1 = *(const v4f*)&snL[sIdx][8 * half + 4];
                        Bh[g].p[4 * p + 0] = pk2(n0[0] * Ug[0], n0[1] * Ug[1]);
                        Bh[g].p[4 * p + 1] = pk2(n0[2] * Ug[2], n0[3] * Ug[3]);
                        Bh[g].p[4 * p + 2] = pk2(n1[0] * Ug[4], n1[1] * Ug[5]);
                        Bh[g].p[4 * p + 3] = pk2(n1[2] * Ug[6], n1[3] * Ug[7]);
                    }
                }

                // f16 GEMM2 for jacobian groups, K=32 (rows w2*32 .. w2*32+31, permuted)
                H16 Ah;
                {
                    int jb = w2 * 32 + 8 * half;       // my half's f16 A columns: 8 at jb, 8 at jb+16
                    const _Float16* ap = &AfL[lane & 3][jb];
                    Ah.q[0] = *(const uint4*)(ap);
                    Ah.q[1] = *(const uint4*)(ap + 16);
                }
                #pragma unroll
                for (int g = 0; g < 8; ++g) {
                    acc[g + 1] = __builtin_amdgcn_wmma_f32_16x16x32_f16(
                        false, Ah.v, false, Bh[g].v, (short)0, acc[g + 1], false, false);
                }
            }

            // ---- spill s_aug (rows 0..3, lanes 0..15 of D) to LDS ----
            if (lane < 16) {
                #pragma unroll
                for (int i = 0; i < HALFD; ++i) {
                    Sg[m][i][lane][0] = acc[0][i];                 // value: N = sample
                    #pragma unroll
                    for (int g = 0; g < 8; ++g) {
                        int s = 2 * g + (lane >> 3);
                        int c = lane & 7;
                        Sg[m][i][s][1 + c] = acc[g + 1][i];        // d s_i / d x_c
                    }
                }
            }
            __syncthreads();
        }

        // ---- coupling update: y2 = x2*exp(s) + t, chain-rule the Jacobian rows ----
        #pragma unroll
        for (int rep = 0; rep < 2; ++rep) {
            int r = (lane >> 4) + 2 * rep;   // 64 (sample,row) tasks over 32 lanes
            int s = l15;
            float xv  = Xs[s][h2 + r];
            float es  = __expf(Sg[0][r][s][0]);
            float tv  = Sg[1][r][s][0];
            float xes = xv * es;
            #pragma unroll
            for (int c = 0; c < DIM; ++c) {
                Js[s][h2 + r][c] = xes * Sg[0][r][s][1 + c]
                                 + es * Js[s][h2 + r][c]
                                 + Sg[1][r][s][1 + c];
            }
            Xs[s][h2 + r] = xes + tv;
        }
        __syncthreads();
    }

    // ---- epilogue: G = J@J ; solve G z = y (unrolled Gauss-Jordan) ; out = -z ----
    if (lane < 16) {
        int s = lane;
        float Jr[DIM][DIM], y[DIM];
        #pragma unroll
        for (int i = 0; i < DIM; ++i) {
            y[i] = Xs[s][i];
            #pragma unroll
            for (int j = 0; j < DIM; ++j) Jr[i][j] = Js[s][i][j];
        }
        float G[DIM][DIM];
        #pragma unroll
        for (int i = 0; i < DIM; ++i) {
            #pragma unroll
            for (int j = 0; j < DIM; ++j) {
                float a = 0.0f;
                #pragma unroll
                for (int t = 0; t < DIM; ++t) a += Jr[i][t] * Jr[t][j];
                G[i][j] = a;
            }
        }
        #pragma unroll
        for (int kk = 0; kk < DIM; ++kk) {
            float ip = 1.0f / G[kk][kk];
            #pragma unroll
            for (int j = 0; j < DIM; ++j) G[kk][j] *= ip;
            y[kk] *= ip;
            #pragma unroll
            for (int i = 0; i < DIM; ++i) {
                if (i == kk) continue;
                float f = G[i][kk];
                #pragma unroll
                for (int j = 0; j < DIM; ++j) G[i][j] -= f * G[kk][j];
                y[i] -= f * y[kk];
            }
        }
        #pragma unroll
        for (int i = 0; i < DIM; ++i)
            out[(size_t)(base + s) * DIM + i] = -y[i];
    }
}

extern "C" void kernel_launch(void* const* d_in, const int* in_sizes, int n_in,
                              void* d_out, int out_size, void* d_ws, size_t ws_size,
                              hipStream_t stream) {
    (void)in_sizes; (void)n_in; (void)out_size; (void)d_ws; (void)ws_size;
    const float* x  = (const float*)d_in[0];
    const float* Ws = (const float*)d_in[1];
    const float* bs = (const float*)d_in[2];
    const float* As = (const float*)d_in[3];
    float* out = (float*)d_out;
    diffeo_dynamics_kernel<<<dim3(BATCH / TS), dim3(32), 0, stream>>>(x, Ws, bs, As, out);
}
